// Set2Vec_23390391894896
// MI455X (gfx1250) — matrix operand order repaired and checked by reading, not compiled, not run on hardware
//
#include <hip/hip_runtime.h>
#include <math.h>

// ---------------------------------------------------------------------------
// CDNA5 (gfx1250) Set2Vec implementation.
// GEMMs: v_wmma_f32_16x16x32_bf16 with double-buffered async global->LDS
// staging of the B tile (ASYNCcnt) and register-pipelined A fragments.
// K-loop unrolled by 2 so the LDS buffer index is static and B-fragment
// ds_loads are batched ahead of the WMMA chain.
// ---------------------------------------------------------------------------

typedef __attribute__((ext_vector_type(16))) __bf16 v16bf;
typedef __attribute__((ext_vector_type(8)))  float  v8f;

#define DEV __device__ __forceinline__

DEV unsigned short f32_to_bf16(float f) {
    unsigned int u = __float_as_uint(f);
    unsigned int r = (u + 0x7FFFu + ((u >> 16) & 1u)) >> 16;   // round-to-nearest-even
    return (unsigned short)r;
}

union Frag { v16bf v; unsigned int u[8]; };

// Raw LDS byte offset of a shared-memory pointer: flat LDS-aperture addresses
// keep the LDS offset in the low 32 bits (ISA 10.2 aperture rules).
DEV unsigned lds_offset(const void* p) { return (unsigned)(unsigned long long)p; }

// ---------------------------------------------------------------------------
// Elementwise prep kernels
// ---------------------------------------------------------------------------
__global__ void f32_to_bf16_kernel(const float* __restrict__ src,
                                   unsigned short* __restrict__ dst, long long n) {
    long long i = (long long)blockIdx.x * blockDim.x + threadIdx.x;
    long long stride = (long long)gridDim.x * blockDim.x;
    for (; i < n; i += stride) dst[i] = f32_to_bf16(src[i]);
}

// Concatenate the 4 gate weights [1024,512] into one bf16 matrix [K=1024, N=2048]:
// columns [0,512)=i, [512,1024)=f, [1024,1536)=c, [1536,2048)=o
__global__ void wcat_kernel(const float* __restrict__ wi, const float* __restrict__ wf,
                            const float* __restrict__ wc, const float* __restrict__ wo,
                            unsigned short* __restrict__ dst) {
    long long i = (long long)blockIdx.x * blockDim.x + threadIdx.x;
    long long n = 1024LL * 512LL;
    long long stride = (long long)gridDim.x * blockDim.x;
    for (; i < n; i += stride) {
        long long k = i >> 9;          // / 512
        long long c = i & 511;         // % 512
        long long row = k * 2048;
        dst[row + c]        = f32_to_bf16(wi[i]);
        dst[row + 512 + c]  = f32_to_bf16(wf[i]);
        dst[row + 1024 + c] = f32_to_bf16(wc[i]);
        dst[row + 1536 + c] = f32_to_bf16(wo[i]);
    }
}

__global__ void zero_kernel(float* __restrict__ p, long long n) {
    long long i = (long long)blockIdx.x * blockDim.x + threadIdx.x;
    long long stride = (long long)gridDim.x * blockDim.x;
    for (; i < n; i += stride) p[i] = 0.0f;
}

__global__ void copy_kernel(const float* __restrict__ s, float* __restrict__ d, long long n) {
    long long i = (long long)blockIdx.x * blockDim.x + threadIdx.x;
    long long stride = (long long)gridDim.x * blockDim.x;
    for (; i < n; i += stride) d[i] = s[i];
}

// ---------------------------------------------------------------------------
// WMMA GEMM: C[M,N] = A_bf16[M,K] @ B_bf16[K,N] (+ bias[N])
// Block = 128 threads = 4 waves; each wave computes a 16x64 output tile.
// B tile [32 x 64] bf16 (4 KB) is shared by all waves: streamed into LDS with
// global_load_async_to_lds_b128 (double-buffered, ASYNCcnt), consumed with
// batched 16-bit ds_loads. A fragments are register double-buffered.
// Fragment layouts per CDNA5 ISA 7.12.2 (wave32):
//   A 16x32 bf16: lane l<16 -> row l, K={0..7,16..23}; lanes 16..31 -> K={8..15,24..31}
//   B 32x16 bf16: lane l -> col l&15, K sets by lane half as for A
//   C 16x16 f32 : VGPR v -> row v + 8*(lane>=16), col = lane&15
// ---------------------------------------------------------------------------
__global__ __launch_bounds__(128) void wmma_gemm_bf16(
    const unsigned short* __restrict__ A,   // [M,K] bf16 (row major)
    const unsigned short* __restrict__ B,   // [K,N] bf16 (row major)
    const float* __restrict__ bias,         // [N] or nullptr
    float* __restrict__ C,                  // [M,N] f32
    int M, int N, int K)
{
    __shared__ unsigned short btile[2][32][64];   // 8 KB double-buffered B tile

    const int tid   = threadIdx.x;
    const int lane  = tid & 31;
    const int wave  = tid >> 5;
    const int half  = lane >> 4;        // 0 or 1
    const int row16 = lane & 15;

    const int mtile  = blockIdx.y * 64 + wave * 16;
    const int ntile0 = blockIdx.x * 64;

    // async-copy mapping: thread t copies 16B chunk (t&7) of rows (t>>3) and (t>>3)+16
    const int brow0  = tid >> 3;        // 0..15
    const int belem  = (tid & 7) * 8;   // bf16 element offset within 64-wide row (16B chunks)

    const unsigned short* gs = B + (long long)brow0 * N + ntile0 + belem;
    const long long ghalf = (long long)16 * N;   // 16 K-rows
    const long long gstep = (long long)32 * N;   // one K-step
    const int nk = K >> 5;               // always even (K = 512 or 1024)

    v8f acc[4];
#pragma unroll
    for (int t = 0; t < 4; t++) acc[t] = {};

    const unsigned short* aptr = A + (long long)(mtile + row16) * K + half * 8;

    // --- helpers -----------------------------------------------------------
    auto issue_async = [&](const unsigned short* src, int p) {
        unsigned l0 = lds_offset(&btile[p][brow0][belem]);
        unsigned l1 = lds_offset(&btile[p][brow0 + 16][belem]);
        asm volatile(
            "global_load_async_to_lds_b128 %0, %2, off\n\t"
            "global_load_async_to_lds_b128 %1, %3, off\n\t"
            "s_wait_xcnt 0x0"
            :: "v"(l0), "v"(l1), "v"(src), "v"(src + ghalf)
            : "memory");
    };
    auto wait_async = [&]() {
        asm volatile("s_wait_asynccnt 0x0" ::: "memory");
    };
    auto load_a = [&](int kt) {
        Frag a;
        const unsigned short* p = aptr + kt * 32;
#pragma unroll
        for (int i = 0; i < 4; i++) a.u[i]     = *(const unsigned int*)(p + 2 * i);
#pragma unroll
        for (int i = 0; i < 4; i++) a.u[4 + i] = *(const unsigned int*)(p + 16 + 2 * i);
        return a;
    };
    // Load all 4 B fragments first (64 16-bit ds_loads batched), then run the
    // 4 WMMAs back-to-back (disjoint accumulators, shared A -> no hazards).
    auto compute = [&](const Frag& a, int p) {
        Frag bf[4];
#pragma unroll
        for (int t = 0; t < 4; t++) {
#pragma unroll
            for (int j = 0; j < 8; j++) {
                const int k0 = (j < 4) ? (8 * half + 2 * j) : (16 + 8 * half + 2 * (j - 4));
                bf[t].u[j] = (unsigned)btile[p][k0][t * 16 + row16] |
                             ((unsigned)btile[p][k0 + 1][t * 16 + row16] << 16);
            }
        }
#pragma unroll
        for (int t = 0; t < 4; t++) {
            acc[t] = __builtin_amdgcn_wmma_f32_16x16x32_bf16(
                false, a.v, false, bf[t].v, (short)0, acc[t], false, false);
        }
    };
    // -----------------------------------------------------------------------

    // prologue: stage tile 0 into buf0, load A fragment 0
    issue_async(gs, 0);
    Frag a = load_a(0);
    wait_async();
    __syncthreads();

    for (int kt = 0; kt < nk; kt += 2) {
        // ---- phase A: consume buf0 (tile kt), stream tile kt+1 -> buf1 ----
        gs += gstep;
        issue_async(gs, 1);
        Frag an = load_a(kt + 1);
        compute(a, 0);
        a = an;
        wait_async();
        __syncthreads();

        // ---- phase B: consume buf1 (tile kt+1), stream tile kt+2 -> buf0 --
        const bool more = (kt + 2 < nk);
        if (more) { gs += gstep; issue_async(gs, 0); }
        Frag an2 = a;
        if (more) an2 = load_a(kt + 2);
        compute(a, 1);
        a = an2;
        if (more) { wait_async(); __syncthreads(); }
    }

#pragma unroll
    for (int t = 0; t < 4; t++) {
        const int cc = ntile0 + t * 16 + row16;
        const float bv = bias ? bias[cc] : 0.0f;
#pragma unroll
        for (int v = 0; v < 8; v++) {
            const int cr = mtile + v + 8 * half;
            C[(long long)cr * N + cc] = acc[t][v] + bv;
        }
    }
}

// ---------------------------------------------------------------------------
// Gate nonlinearities + LSTM cell update.
// gates[b, 0:512)=i, [512:1024)=f, [1024:1536)=c~, [1536:2048)=o
// ---------------------------------------------------------------------------
__global__ __launch_bounds__(256) void gates_kernel(
    const float* __restrict__ g,
    const float* __restrict__ b_i, const float* __restrict__ b_f,
    const float* __restrict__ b_c, const float* __restrict__ b_o,
    float* __restrict__ c_state, float* __restrict__ m_f32,
    unsigned short* __restrict__ ml_bf16)
{
    const int idx = blockIdx.x * blockDim.x + threadIdx.x;   // [0, 512*512)
    const int b = idx >> 9;
    const int j = idx & 511;
    const float* gr = g + (long long)b * 2048;

    const float it = 1.0f / (1.0f + __expf(-(gr[j]        + b_i[j])));
    const float ft = 1.0f / (1.0f + __expf(-(gr[512 + j]  + b_f[j])));
    const float gt = tanhf(gr[1024 + j] + b_c[j]);
    const float ot = 1.0f / (1.0f + __expf(-(gr[1536 + j] + b_o[j])));

    const float cn = ft * c_state[idx] + it * gt;
    c_state[idx] = cn;
    const float ml = ot * tanhf(cn);
    m_f32[(long long)b * 1024 + j] = ml;
    ml_bf16[idx] = f32_to_bf16(ml);
}

// ---------------------------------------------------------------------------
// Fused attention: energies -> softmax(N) -> weighted read.
// One wave per (b,h); lane owns 2 of the 64 Dh values.
// ---------------------------------------------------------------------------
__global__ __launch_bounds__(256) void attn_kernel(
    const float* __restrict__ hs,       // [B,128,8,64] = [B,128,512]
    const float* __restrict__ query,    // [B,512]
    const float* __restrict__ att_v,    // [8,64]
    const float* __restrict__ mask,     // [B,128]
    float* __restrict__ m_f32)          // [B,1024]
{
    __shared__ float e_sm[8][128];
    const int lane = threadIdx.x & 31;
    const int wave = threadIdx.x >> 5;
    const int unit = blockIdx.x * 8 + wave;   // (b,h)
    const int b = unit >> 3;
    const int h = unit & 7;

    const float* hsb = hs + (long long)b * 128 * 512 + h * 64;   // row stride 512
    const float2 q = ((const float2*)(query + (long long)b * 512 + h * 64))[lane];
    const float2 v = ((const float2*)(att_v + h * 64))[lane];

    // Pass 1: energies
    for (int n = 0; n < 128; n++) {
        __builtin_prefetch(hsb + (n + 2) * 512 + lane * 2, 0, 1);  // global_prefetch_b8
        const float2 x = ((const float2*)(hsb + n * 512))[lane];
        float ep = tanhf(q.x + x.x) * v.x + tanhf(q.y + x.y) * v.y;
#pragma unroll
        for (int o = 16; o >= 1; o >>= 1) ep += __shfl_xor(ep, o, 32);
        if (lane == 0)
            e_sm[wave][n] = ep + (1.0f - mask[b * 128 + n]) * (-1e10f);
    }
    __syncthreads();

    // Softmax over N=128 (4 values per lane)
    float ex[4];
    float mx = -3.4e38f;
#pragma unroll
    for (int j = 0; j < 4; j++) mx = fmaxf(mx, e_sm[wave][lane + 32 * j]);
#pragma unroll
    for (int o = 16; o >= 1; o >>= 1) mx = fmaxf(mx, __shfl_xor(mx, o, 32));
    float sum = 0.0f;
#pragma unroll
    for (int j = 0; j < 4; j++) { ex[j] = __expf(e_sm[wave][lane + 32 * j] - mx); sum += ex[j]; }
#pragma unroll
    for (int o = 16; o >= 1; o >>= 1) sum += __shfl_xor(sum, o, 32);
    const float inv = 1.0f / sum;
#pragma unroll
    for (int j = 0; j < 4; j++) e_sm[wave][lane + 32 * j] = ex[j] * inv;
    __syncthreads();

    // Pass 2: weighted read
    float2 acc; acc.x = 0.0f; acc.y = 0.0f;
    for (int n = 0; n < 128; n++) {
        const float a = e_sm[wave][n];
        const float2 x = ((const float2*)(hsb + n * 512))[lane];
        acc.x += a * x.x;
        acc.y += a * x.y;
    }
    ((float2*)(m_f32 + (long long)b * 1024 + 512 + h * 64))[lane] = acc;
}

// ---------------------------------------------------------------------------
// Host launcher
// ---------------------------------------------------------------------------
extern "C" void kernel_launch(void* const* d_in, const int* in_sizes, int n_in,
                              void* d_out, int out_size, void* d_ws, size_t ws_size,
                              hipStream_t stream)
{
    const float* hidden     = (const float*)d_in[0];
    const float* mask       = (const float*)d_in[1];
    const float* fc_w       = (const float*)d_in[2];
    const float* fc_b       = (const float*)d_in[3];
    const float* w_im       = (const float*)d_in[4];
    const float* b_i        = (const float*)d_in[5];
    const float* w_fm       = (const float*)d_in[6];
    const float* b_f        = (const float*)d_in[7];
    const float* w_cm       = (const float*)d_in[8];
    const float* b_c        = (const float*)d_in[9];
    const float* w_om       = (const float*)d_in[10];
    const float* b_o        = (const float*)d_in[11];
    const float* m_to_query = (const float*)d_in[12];
    const float* att_v      = (const float*)d_in[13];

    const int B = 512, N = 128, D = 512, H = 8;
    const long long MN = (long long)B * N;          // 65536 rows

    // Workspace carve-out (256B aligned slices)
    char* ws = (char*)d_ws;
    auto carve = [&](size_t bytes) -> char* {
        char* p = ws;
        ws += (bytes + 255) & ~(size_t)255;
        return p;
    };
    unsigned short* A_bf16  = (unsigned short*)carve((size_t)MN * D * 2);       // 64 MB
    float*          hs      = (float*)carve((size_t)MN * D * 4);                // 128 MB
    unsigned short* fc_wb   = (unsigned short*)carve((size_t)D * D * 2);
    unsigned short* wcat    = (unsigned short*)carve(1024ULL * 2048ULL * 2);
    unsigned short* mtq     = (unsigned short*)carve((size_t)D * D * 2);
    float*          m_f32   = (float*)carve((size_t)B * 2 * D * 4);
    unsigned short* m_bf16  = (unsigned short*)carve((size_t)B * 2 * D * 2);
    float*          c_state = (float*)carve((size_t)B * D * 4);
    float*          gates   = (float*)carve((size_t)B * 4 * D * 4);
    unsigned short* ml_bf16 = (unsigned short*)carve((size_t)B * D * 2);
    float*          query   = (float*)carve((size_t)B * D * 4);
    (void)ws_size; (void)in_sizes; (void)n_in; (void)out_size;

    // --- prep: bf16 conversions + weight concat, init state ---
    f32_to_bf16_kernel<<<4096, 256, 0, stream>>>(hidden, A_bf16, MN * D);
    f32_to_bf16_kernel<<<512, 256, 0, stream>>>(fc_w, fc_wb, (long long)D * D);
    f32_to_bf16_kernel<<<512, 256, 0, stream>>>(m_to_query, mtq, (long long)D * D);
    wcat_kernel<<<1024, 256, 0, stream>>>(w_im, w_fm, w_cm, w_om, wcat);
    zero_kernel<<<1024, 256, 0, stream>>>(m_f32, (long long)B * 2 * D);
    zero_kernel<<<512, 256, 0, stream>>>(c_state, (long long)B * D);

    // --- input projection: hs = hidden @ fc_w + fc_b  (65536 x 512 x 512) ---
    wmma_gemm_bf16<<<dim3(D / 64, (int)(MN / 64)), 128, 0, stream>>>(
        A_bf16, fc_wb, fc_b, hs, (int)MN, D, D);

    // --- 6 recurrent steps ---
    for (int s = 0; s < 6; s++) {
        // m -> bf16 for gate GEMM
        f32_to_bf16_kernel<<<512, 256, 0, stream>>>(m_f32, m_bf16, (long long)B * 2 * D);
        // all 4 gates in one GEMM: [512,1024] x [1024,2048]
        wmma_gemm_bf16<<<dim3(2048 / 64, B / 64), 128, 0, stream>>>(
            m_bf16, wcat, nullptr, gates, B, 2048, 1024);
        // LSTM cell update (+ bf16 m_lstm for query GEMM)
        gates_kernel<<<(B * D) / 256, 256, 0, stream>>>(
            gates, b_i, b_f, b_c, b_o, c_state, m_f32, ml_bf16);
        // query = m_lstm @ m_to_query : [512,512] x [512,512]
        wmma_gemm_bf16<<<dim3(D / 64, B / 64), 128, 0, stream>>>(
            ml_bf16, mtq, nullptr, query, B, D, D);
        // attention read -> m_f32[:, 512:1024)
        attn_kernel<<<(B * H) / 8, 256, 0, stream>>>(hs, query, att_v, mask, m_f32);
    }

    // --- output: final m [512, 1024] ---
    copy_kernel<<<1024, 256, 0, stream>>>(m_f32, (float*)d_out, (long long)B * 2 * D);
}
